// Multi_Headed_86912958201908
// MI455X (gfx1250) — compile-verified, matrix operand order checked by reading, not run on hardware
//
#include <hip/hip_runtime.h>
#include <hip/hip_bf16.h>
#include <cstdint>

// ---------------------------------------------------------------------------
// Multi-head causal attention for MI455X (gfx1250), wave32 + WMMA bf16.
// B=8, S=2048, E=768, H=3, D=256.
//
// Pipeline:
//   0) pack_w_kernel x4 : weights f32 -> bf16, pre-swizzled into exact WMMA
//                         B-fragment order (one coalesced 32B load / fragment)
//   1) proj_kernel   x3 : X(f32) x Wpacked -> bf16 Q/K [B,H,S,D]; V written
//                         directly in packed PV B-fragment order
//   2) attn_kernel      : flash attention, 1 wave / 16-row q-tile, online
//                         softmax, P transposed via LDS, heads -> bf16 [B,S,E]
//   3) outproj_kernel   : concat(bf16) x Wo(packed bf16) -> f32 out
// All matmuls: v_wmma_f32_16x16x32_bf16, f32 accumulation, 16x64 per wave.
// K-loops fully unrolled with literal instruction offsets (no address VALU).
// ---------------------------------------------------------------------------

#define BB 8
#define SS 2048
#define EE 768
#define HH 3
#define DD 256

typedef __bf16 bf16_t;
typedef __bf16 v16bf __attribute__((ext_vector_type(16)));
typedef __bf16 v8bf  __attribute__((ext_vector_type(8)));
typedef float  v8f   __attribute__((ext_vector_type(8)));
typedef float  v8fv  __attribute__((ext_vector_type(8)));

// K-offset pattern for 16-bit WMMA fragments (ISA 7.12.2):
// lanes 0-15 hold K = {0..7, 16..23}, lanes 16-31 hold K = {8..15, 24..31}.
__device__ __forceinline__ int frag_k(int pair, int laneHalf) {
  return 2 * (pair & 3) + 16 * (pair >> 2) + 8 * laneHalf;
}

// Fragment from a per-lane base pointer (already includes row*ld + 8*laneHalf)
// plus a compile-time element offset: two aligned 16B loads.
__device__ __forceinline__ v16bf frag_bf16(const bf16_t* __restrict__ p) {
  v8bf lo = *(const v8bf*)(p);
  v8bf hi = *(const v8bf*)(p + 16);
  v16bf f;
#pragma unroll
  for (int i = 0; i < 8; ++i) { f[i] = lo[i]; f[i + 8] = hi[i]; }
  return f;
}

__device__ __forceinline__ v16bf frag_f32(const float* __restrict__ p) {
  v8fv lo = *(const v8fv*)(p);
  v8fv hi = *(const v8fv*)(p + 16);
  v16bf f;
#pragma unroll
  for (int i = 0; i < 8; ++i) {
    f[i]     = (bf16_t)lo[i];      // v_cvt_pk_bf16_f32
    f[i + 8] = (bf16_t)hi[i];
  }
  return f;
}

__device__ __forceinline__ v8f wmma_bf16(v16bf a, v16bf b, v8f c) {
  return __builtin_amdgcn_wmma_f32_16x16x32_bf16(false, a, false, b,
                                                 (short)0, c, false, false);
}

// ---------------------------------------------------------------------------
// Kernel 0: pack a row-major f32 [K x N] matrix (nMat of them) into bf16
// B-fragment order: dst[mat][nt][ke][lane][i] = src[mat][e][d] with
// e = ke*32 + frag_k(i/2, lane/16) + (i&1), d = nt*16 + (lane&15).
// ---------------------------------------------------------------------------
__global__ __launch_bounds__(256) void pack_w_kernel(const float* __restrict__ src,
                                                     bf16_t* __restrict__ dst,
                                                     int Kdim, int Ndim,
                                                     int ntCount, int nMat) {
  const int kSteps = Kdim >> 5;
  int idx = blockIdx.x * 256 + threadIdx.x;
  int total = nMat * ntCount * kSteps * 512;
  if (idx >= total) return;
  int i    = idx & 15;
  int lane = (idx >> 4) & 31;
  int rest = idx >> 9;
  int ke  = rest % kSteps;  rest /= kSteps;
  int nt  = rest % ntCount;
  int mat = rest / ntCount;
  int half = lane >> 4, n = lane & 15;
  int e = ke * 32 + frag_k(i >> 1, half) + (i & 1);
  int d = nt * 16 + n;
  dst[idx] = (bf16_t)src[((size_t)mat * Kdim + e) * Ndim + d];
}

// ---------------------------------------------------------------------------
// Kernel 1: per-head projection, 16x64 output tile per wave, K fully
// unrolled (24 steps x 4 WMMAs, A reused 4x, all offsets literal).
// ---------------------------------------------------------------------------
__global__ __launch_bounds__(128) void proj_kernel(const float* __restrict__ X,
                                                   const bf16_t* __restrict__ Wp,
                                                   bf16_t* __restrict__ Out,
                                                   float scale, int packV) {
  const int lane = threadIdx.x & 31;
  const int laneHalf = lane >> 4;
  const int nl = lane & 15;

  int tile = blockIdx.x * 4 + (threadIdx.x >> 5);
  int dt = tile & 3;        tile >>= 2;   // D/64 = 4 col-groups
  int st = tile & 127;      tile >>= 7;   // S/16 = 128 row tiles
  int h  = tile % HH;
  int b  = tile / HH;
  const int s0 = st * 16, d0 = dt * 64;

  // Per-wave/lane base pointers; everything below uses literal offsets.
  const float*  pA = X + (size_t)b * SS * EE + (size_t)(s0 + nl) * EE + 8 * laneHalf;
  const bf16_t* pB = Wp + (size_t)h * (16 * 24 * 512)
                        + (size_t)(d0 >> 4) * (24 * 512) + (size_t)lane * 16;

  v8f acc[4] = {};
#pragma unroll
  for (int ke = 0; ke < 24; ++ke) {
    v16bf a = frag_f32(pA + ke * 32);
#pragma unroll
    for (int j = 0; j < 4; ++j) {
      v16bf bb = *(const v16bf*)(pB + j * (24 * 512) + ke * 512);
      acc[j] = wmma_bf16(a, bb, acc[j]);
    }
  }

  bf16_t* O = Out + ((size_t)b * HH + h) * SS * DD;
  if (!packV) {
#pragma unroll
    for (int j = 0; j < 4; ++j)
#pragma unroll
      for (int r = 0; r < 8; ++r) {
        int row = s0 + r + 8 * laneHalf;
        O[(size_t)row * DD + d0 + 16 * j + nl] = (bf16_t)(acc[j][r] * scale);
      }
  } else {
    // scatter into PV B-fragment order (inverse of frag_k mapping)
#pragma unroll
    for (int j = 0; j < 4; ++j)
#pragma unroll
      for (int r = 0; r < 8; ++r) {
        int s    = s0 + r + 8 * laneHalf;
        int kb32 = s >> 5, srel = s & 31;
        int hp   = (srel >> 3) & 1;
        int pp   = ((srel & 7) >> 1) + 4 * (srel >> 4);
        int ii   = 2 * pp + (srel & 1);
        int lanep = hp * 16 + nl;
        int tt   = (d0 >> 4) + j;
        O[(((size_t)kb32 * 16 + tt) * 32 + lanep) * 16 + ii] = (bf16_t)acc[j][r];
      }
  }
}

// ---------------------------------------------------------------------------
// Kernel 2: flash attention. One wave (block of 32) per (b,h,16-row q-tile).
// ---------------------------------------------------------------------------
__global__ __launch_bounds__(32) void attn_kernel(const bf16_t* __restrict__ Q,
                                                  const bf16_t* __restrict__ K,
                                                  const bf16_t* __restrict__ Vp,
                                                  bf16_t* __restrict__ C) {
  const int lane = threadIdx.x & 31;
  const int laneHalf = lane >> 4;
  const int nl = lane & 15;

  int t = blockIdx.x;
  int st = t & 127;  t >>= 7;
  int h  = t % HH;
  int b  = t / HH;
  const int q0 = st * 16;

  const bf16_t* Km  = K  + ((size_t)b * HH + h) * SS * DD;
  const bf16_t* VpH = Vp + ((size_t)b * HH + h) * SS * DD;

  // Q tile: 16 rows x 256 d -> 8 A-fragments in VGPRs (literal offsets).
  const bf16_t* pQ = Q + ((size_t)b * HH + h) * SS * DD
                       + (size_t)(q0 + nl) * DD + 8 * laneHalf;
  v16bf qf[8];
#pragma unroll
  for (int j = 0; j < 8; ++j) qf[j] = frag_bf16(pQ + 32 * j);

  v8f o[16];
#pragma unroll
  for (int tt = 0; tt < 16; ++tt) o[tt] = (v8f){};

  float m[8], l[8];
#pragma unroll
  for (int r = 0; r < 8; ++r) { m[r] = -__builtin_inff(); l[r] = 0.0f; }

  __shared__ bf16_t Pst[16 * 32];

  // Hoisted streaming pointers (single 64-bit bump per 32-key step).
  const bf16_t* pK = Km + (size_t)nl * DD + 8 * laneHalf;
  const bf16_t* pV = VpH + (size_t)lane * 16;

  const int kend = q0 + 16;                 // causal: keys [0, q0+16)
  for (int kb = 0; kb < kend; kb += 32) {
    if (kb + 32 < kend) {                   // prefetch next K block (gfx1250)
      __builtin_prefetch(pK + (size_t)32 * DD, 0, 1);
    }

    // ---- scores: S[16x32] = Q(16x256) . K^T(256x32) ----
    v8f s0t = {}, s1t = {};
#pragma unroll
    for (int j = 0; j < 8; ++j) {
      v16bf kf0 = frag_bf16(pK + 32 * j);
      v16bf kf1 = frag_bf16(pK + 16 * DD + 32 * j);
      s0t = wmma_bf16(qf[j], kf0, s0t);
      s1t = wmma_bf16(qf[j], kf1, s1t);
    }

    // ---- causal mask (only near the diagonal) ----
    if (kb + 31 > q0) {
#pragma unroll
      for (int r = 0; r < 8; ++r) {
        int row = q0 + r + 8 * laneHalf;
        if (kb + nl      > row) s0t[r] = -__builtin_inff();
        if (kb + 16 + nl > row) s1t[r] = -__builtin_inff();
      }
    }

    // ---- online softmax ----
    float alpha[8];
#pragma unroll
    for (int r = 0; r < 8; ++r) {
      float v = fmaxf(s0t[r], s1t[r]);
      v = fmaxf(v, __shfl_xor(v, 1, 16));
      v = fmaxf(v, __shfl_xor(v, 2, 16));
      v = fmaxf(v, __shfl_xor(v, 4, 16));
      v = fmaxf(v, __shfl_xor(v, 8, 16));
      float mn = fmaxf(m[r], v);
      alpha[r] = __expf(m[r] - mn);
      m[r] = mn;
    }
#pragma unroll
    for (int r = 0; r < 8; ++r) {
      float p0 = __expf(s0t[r] - m[r]);
      float p1 = __expf(s1t[r] - m[r]);
      float rs = p0 + p1;
      rs += __shfl_xor(rs, 1, 16);
      rs += __shfl_xor(rs, 2, 16);
      rs += __shfl_xor(rs, 4, 16);
      rs += __shfl_xor(rs, 8, 16);
      l[r] = l[r] * alpha[r] + rs;
      s0t[r] = p0; s1t[r] = p1;
    }

    // ---- transpose P through LDS into an A-fragment ----
    __syncthreads();
#pragma unroll
    for (int r = 0; r < 8; ++r) {
      int row = r + 8 * laneHalf;
      Pst[row * 32 + nl]      = (bf16_t)s0t[r];
      Pst[row * 32 + 16 + nl] = (bf16_t)s1t[r];
    }
    __syncthreads();
    v16bf pf = frag_bf16(Pst + (size_t)nl * 32 + 8 * laneHalf);

    // ---- rescale accumulators; O += P(16x32) . V(32x256) ----
#pragma unroll
    for (int tt = 0; tt < 16; ++tt)
#pragma unroll
      for (int r = 0; r < 8; ++r) o[tt][r] *= alpha[r];

#pragma unroll
    for (int tt = 0; tt < 16; ++tt) {
      v16bf vf = *(const v16bf*)(pV + tt * 512);   // packed: one 32B load
      o[tt] = wmma_bf16(pf, vf, o[tt]);
    }

    pK += (size_t)32 * DD;       // next 32 keys
    pV += 16 * 512;              // next packed 32-key group
  }

  // ---- normalize, write concat[b, q0+row, h*D + d] ----
  bf16_t* Cb = C + (size_t)b * SS * EE;
#pragma unroll
  for (int r = 0; r < 8; ++r) {
    int row = q0 + r + 8 * laneHalf;
    float inv = 1.0f / l[r];
#pragma unroll
    for (int tt = 0; tt < 16; ++tt)
      Cb[(size_t)row * EE + h * DD + 16 * tt + nl] = (bf16_t)(o[tt][r] * inv);
  }
}

// ---------------------------------------------------------------------------
// Kernel 3: output projection, 16x64 per wave, K fully unrolled.
// ---------------------------------------------------------------------------
__global__ __launch_bounds__(128) void outproj_kernel(const bf16_t* __restrict__ Cc,
                                                      const bf16_t* __restrict__ Wpo,
                                                      float* __restrict__ Out) {
  const int lane = threadIdx.x & 31;
  const int laneHalf = lane >> 4;
  const int nl = lane & 15;

  int tile = blockIdx.x * 4 + (threadIdx.x >> 5);
  int nt4 = tile % 12;                  // E/64 = 12 col-groups
  int mt  = tile / 12;                  // BS/16 = 1024 row tiles
  const int m0 = mt * 16, n0 = nt4 * 64;

  const bf16_t* pA = Cc + (size_t)(m0 + nl) * EE + 8 * laneHalf;
  const bf16_t* pB = Wpo + (size_t)(n0 >> 4) * (24 * 512) + (size_t)lane * 16;

  v8f acc[4] = {};
#pragma unroll
  for (int ke = 0; ke < 24; ++ke) {
    v16bf a = frag_bf16(pA + ke * 32);
#pragma unroll
    for (int j = 0; j < 4; ++j) {
      v16bf bb = *(const v16bf*)(pB + j * (24 * 512) + ke * 512);
      acc[j] = wmma_bf16(a, bb, acc[j]);
    }
  }

#pragma unroll
  for (int j = 0; j < 4; ++j)
#pragma unroll
    for (int r = 0; r < 8; ++r) {
      int row = m0 + r + 8 * laneHalf;
      Out[(size_t)row * EE + n0 + 16 * j + nl] = acc[j][r];
    }
}

// ---------------------------------------------------------------------------
extern "C" void kernel_launch(void* const* d_in, const int* in_sizes, int n_in,
                              void* d_out, int out_size, void* d_ws, size_t ws_size,
                              hipStream_t stream) {
  const float* Xk = (const float*)d_in[0];
  const float* Xv = (const float*)d_in[1];
  const float* Xq = (const float*)d_in[2];
  const float* Wq = (const float*)d_in[3];
  const float* Wk = (const float*)d_in[4];
  const float* Wv = (const float*)d_in[5];
  const float* Wo = (const float*)d_in[6];
  float* Out = (float*)d_out;

  const size_t BHSD  = (size_t)BB * HH * SS * DD;   // 12,582,912
  const size_t WPACK = (size_t)HH * 16 * 24 * 512;  // 589,824 per QKV weight

  bf16_t* Qb  = (bf16_t*)d_ws;
  bf16_t* Kb  = Qb + BHSD;
  bf16_t* Vp  = Kb + BHSD;        // V in packed PV-fragment order
  bf16_t* Cb  = Vp + BHSD;        // head concat [B,S,E]
  bf16_t* Wpq = Cb + BHSD;
  bf16_t* Wpk = Wpq + WPACK;
  bf16_t* Wpv = Wpk + WPACK;
  bf16_t* Wpo = Wpv + WPACK;

  // 0) pack weights once into fragment order (bf16)
  const int pb = (int)((WPACK + 255) / 256);        // 2304 blocks
  pack_w_kernel<<<pb, 256, 0, stream>>>(Wq, Wpq, EE, DD, 16, HH);
  pack_w_kernel<<<pb, 256, 0, stream>>>(Wk, Wpk, EE, DD, 16, HH);
  pack_w_kernel<<<pb, 256, 0, stream>>>(Wv, Wpv, EE, DD, 16, HH);
  pack_w_kernel<<<pb, 256, 0, stream>>>(Wo, Wpo, EE, EE, 48, 1);

  // 1) projections: B*H*(S/16)*(D/64) = 12288 waves, 4/block
  const int projBlocks = (BB * HH * (SS / 16) * (DD / 64)) / 4;   // 3072
  const float qscale = 0.0625f;                                  // 1/sqrt(256)
  proj_kernel<<<projBlocks, 128, 0, stream>>>(Xq, Wpq, Qb, qscale, 0);
  proj_kernel<<<projBlocks, 128, 0, stream>>>(Xk, Wpk, Kb, 1.0f,   0);
  proj_kernel<<<projBlocks, 128, 0, stream>>>(Xv, Wpv, Vp, 1.0f,   1);

  // 2) flash attention: one wave per (b,h,q-tile)
  const int attnBlocks = BB * HH * (SS / 16);                     // 3072
  attn_kernel<<<attnBlocks, 32, 0, stream>>>(Qb, Kb, Vp, Cb);

  // 3) output projection: (BS/16)*(E/64) = 12288 waves, 4/block
  const int opBlocks = ((BB * SS / 16) * (EE / 64)) / 4;          // 3072
  outproj_kernel<<<opBlocks, 128, 0, stream>>>(Cb, Wpo, Out);

  (void)in_sizes; (void)n_in; (void)out_size; (void)ws_size;
}
